// EMambFlow_70695161692746
// MI455X (gfx1250) — matrix operand order repaired and checked by reading, not compiled, not run on hardware
//
#include <hip/hip_runtime.h>
#include <stdint.h>

// ---------------- types ----------------
typedef __attribute__((ext_vector_type(16))) __bf16 v16bf;
typedef __attribute__((ext_vector_type(8)))  float  v8f;
typedef __attribute__((ext_vector_type(4)))  unsigned int v4u;
typedef __attribute__((ext_vector_type(8)))  int v8i;
typedef __attribute__((ext_vector_type(4)))  int v4i;

#if defined(__HIP_DEVICE_COMPILE__) && __has_builtin(__builtin_amdgcn_tensor_load_to_lds)
#define HAVE_TDM 1
#else
#define HAVE_TDM 0
#endif

// ---------------- device helpers ----------------
__device__ __forceinline__ float d_silu(float x) { return x / (1.0f + __expf(-x)); }
__device__ __forceinline__ float d_softplus(float x) { return (x > 20.0f) ? x : log1pf(__expf(x)); }
__device__ __forceinline__ int   dmin(int a, int b) { return a < b ? a : b; }

// round-to-nearest-even f32 -> bf16
__device__ __forceinline__ __bf16 f2bf(float f) {
  uint32_t u = __builtin_bit_cast(uint32_t, f);
  uint32_t r = u + 0x7FFFu + ((u >> 16) & 1u);
  uint16_t h = (uint16_t)(r >> 16);
  __bf16 o; __builtin_memcpy(&o, &h, 2); return o;
}

// low 32 bits of a generic pointer into LDS are the LDS byte offset
// (CDNA5 ISA)
__device__ __forceinline__ uint32_t lds_off_of(const void* p) {
  return (uint32_t)(uintptr_t)p;
}

#if HAVE_TDM
// Issue a 2D TDM tile load (global f32 -> LDS). D# per CDNA5 ISA.
// tile = tile_x (contiguous) x tile_y rows; OOB vs (tensor_x, tensor_y) zero-fills LDS.
__device__ __forceinline__ void tdm_load_2d(uint32_t lds_off, const void* gptr,
                                            uint32_t tile_x, uint32_t tile_y,
                                            uint32_t tensor_x, uint32_t tensor_y,
                                            uint32_t stride_x) {
  uint64_t ga = (uint64_t)(uintptr_t)gptr;
  v4u g0;
  g0[0] = 1u;                                            // count=1, user descriptor
  g0[1] = lds_off;                                       // lds_addr  [63:32]
  g0[2] = (uint32_t)ga;                                  // global_addr [95:64]
  g0[3] = (uint32_t)((ga >> 32) & 0x01FFFFFFu) | (2u << 30); // addr[56:32], type=2
  v8i g1;
  g1[0] = (int)(2u << 16);                               // data_size = 4B, no multicast
  g1[1] = (int)((tensor_x & 0xFFFFu) << 16);             // tensor_dim0 -> bits 79:48
  g1[2] = (int)((tensor_x >> 16) | ((tensor_y & 0xFFFFu) << 16)); // + tensor_dim1 lo
  g1[3] = (int)((tensor_y >> 16) | (tile_x << 16));      // tensor_dim1 hi, tile_dim0
  g1[4] = (int)(tile_y & 0xFFFFu);                       // tile_dim1 (tile_dim2 = 0)
  g1[5] = (int)stride_x;                                 // tensor_dim0_stride lo
  g1[6] = 0;
  g1[7] = 0;
  v4i z4 = {0, 0, 0, 0};
#if __clang_major__ >= 23
  v8i z8 = {0, 0, 0, 0, 0, 0, 0, 0};
  __builtin_amdgcn_tensor_load_to_lds(g0, g1, z4, z4, z8, 0);
#else
  __builtin_amdgcn_tensor_load_to_lds(g0, g1, z4, z4, 0);
#endif
}
#endif

// cooperative fallback tile load, zero-filled at edges
__device__ __forceinline__ void coop_tile(float* dst, const float* __restrict__ src,
                                          int rows, int vr, int vc, long stride, int tid) {
  for (int i = tid; i < rows * 32; i += 256) {
    int r = i >> 5, c = i & 31;
    dst[i] = (r < vr && c < vc) ? src[(long)r * stride + c] : 0.0f;
  }
}

// ---------------- WMMA GEMM:  C[M,N] = act(scale*(A[M,K] @ B[N,K]^T + bias)) ----------------
// act: 0=none 1=silu 2=relu 3=softplus.  A row stride lda, C row stride ldc, B row stride == K.
#define BM 128
#define BN 64
#define BK 32

__global__ void __launch_bounds__(256)
k_wmma_gemm(const float* __restrict__ A, const float* __restrict__ B,
            const float* __restrict__ bias, float* __restrict__ C,
            int M, int N, int K, int lda, int ldc, int act, float scale) {
  __shared__ float As[2][BM * BK];
  __shared__ float Bs[2][BN * BK];

  const int tid  = threadIdx.x;
  const int lane = tid & 31;
  const int wave = tid >> 5;
  const int la   = lane & 15;
  const int lh   = lane >> 4;
  const int wm   = (wave & 3) * 32;
  const int wn   = (wave >> 2) * 32;
  const long bm  = (long)blockIdx.y * BM;
  const long bn  = (long)blockIdx.x * BN;

  v8f acc[2][2] = {};
  const int nk = (K + BK - 1) / BK;

  auto issue = [&](int buf, int k0) {
#if HAVE_TDM
    if (tid == 0) {
      tdm_load_2d(lds_off_of(&As[buf][0]), A + bm * (long)lda + k0, BK, BM,
                  (uint32_t)(K - k0), (uint32_t)(M - bm), (uint32_t)lda);
      tdm_load_2d(lds_off_of(&Bs[buf][0]), B + bn * (long)K + k0, BK, BN,
                  (uint32_t)(K - k0), (uint32_t)(N - bn), (uint32_t)K);
    }
#else
    coop_tile(&As[buf][0], A + bm * (long)lda + k0, BM,
              dmin(BM, (int)(M - bm)), dmin(BK, K - k0), lda, tid);
    coop_tile(&Bs[buf][0], B + bn * (long)K + k0, BN,
              dmin(BN, (int)(N - bn)), dmin(BK, K - k0), K, tid);
#endif
  };

  issue(0, 0);
#if HAVE_TDM
  if (tid == 0) __builtin_amdgcn_s_wait_tensorcnt(0);
#endif
  __syncthreads();
  // TDM wrote LDS behind the compiler's back: force it to treat LDS as modified
  asm volatile("" ::: "memory");

  int buf = 0;
  for (int kt = 0; kt < nk; ++kt) {
    if (kt + 1 < nk) issue(buf ^ 1, (kt + 1) * BK);   // overlap next DMA with WMMA

    const float* as = &As[buf][0];
    const float* bs = &Bs[buf][0];

    // A fragments: lane la holds row, lh selects K run of 16
    v16bf afr[2];
#pragma unroll
    for (int mi = 0; mi < 2; ++mi) {
      const float* ar = as + (wm + mi * 16 + la) * BK;
      v16bf a{};
#pragma unroll
      for (int i = 0; i < 8; ++i) {
        a[i]     = f2bf(ar[lh * 16 + i]);
        a[i + 8] = f2bf(ar[lh * 16 + 8 + i]);
      }
      afr[mi] = a;
    }
    // B fragments: lane la holds column, lh selects K run of 16
    v16bf bfr[2];
#pragma unroll
    for (int ni = 0; ni < 2; ++ni) {
      const float* br = bs + (wn + ni * 16 + la) * BK + lh * 16;
      v16bf b{};
#pragma unroll
      for (int i = 0; i < 16; ++i) b[i] = f2bf(br[i]);
      bfr[ni] = b;
    }
#pragma unroll
    for (int mi = 0; mi < 2; ++mi)
#pragma unroll
      for (int ni = 0; ni < 2; ++ni)
        acc[mi][ni] = __builtin_amdgcn_wmma_f32_16x16x32_bf16(
            false, afr[mi], false, bfr[ni], (short)0, acc[mi][ni], false, false);

#if HAVE_TDM
    if (tid == 0) __builtin_amdgcn_s_wait_tensorcnt(0);
#endif
    __syncthreads();
    asm volatile("" ::: "memory");
    buf ^= 1;
  }

  // C/D layout: VGPR r -> M = r + lh*8, N = la (per CDNA5 ISA §7.12.2)
#pragma unroll
  for (int mi = 0; mi < 2; ++mi)
#pragma unroll
    for (int ni = 0; ni < 2; ++ni) {
      long col = bn + wn + ni * 16 + la;
#pragma unroll
      for (int r = 0; r < 8; ++r) {
        long row = bm + wm + mi * 16 + lh * 8 + r;
        if (row < M && col < N) {
          float v = acc[mi][ni][r] * scale;
          if (bias) v += bias[col] * scale;
          if (act == 1) v = d_silu(v);
          else if (act == 2) v = fmaxf(v, 0.0f);
          else if (act == 3) v = d_softplus(v);
          C[row * (long)ldc + col] = v;
        }
      }
    }
}

// ---------------- elementwise / support kernels ----------------
#define GS_LOOP(i, total) \
  for (size_t i = (size_t)blockIdx.x * blockDim.x + threadIdx.x; i < (total); \
       i += (size_t)gridDim.x * blockDim.x)

__global__ void k_pack_patch(const float* __restrict__ x, float* __restrict__ Ap,
                             int C, int T, int H, int W, int kt, int ps,
                             int To, int Ho, int Wo, size_t total) {
  size_t K = (size_t)C * kt * ps * ps;
  GS_LOOP(i, total) {
    size_t m = i / K, k = i % K;
    int q = (int)(k % ps);
    int p = (int)((k / ps) % ps);
    int kk = (int)((k / ((size_t)ps * ps)) % kt);
    int c = (int)(k / ((size_t)ps * ps * kt));
    int w = (int)(m % Wo);
    int h = (int)((m / Wo) % Ho);
    int t = (int)(m / ((size_t)Wo * Ho));
    Ap[i] = x[(((size_t)c * T + (size_t)t * kt + kk) * H + (size_t)h * ps + p) * W +
              (size_t)w * ps + q];
  }
}

__global__ void k_add_tpos(float* __restrict__ seq, const float* __restrict__ tpos,
                           int HW, int E, size_t total) {
  GS_LOOP(i, total) {
    int e = (int)(i % E);
    int t = (int)(i / ((size_t)HW * E));
    seq[i] += tpos[(size_t)t * E + e];
  }
}

__global__ void k_add(const float* __restrict__ a, const float* __restrict__ b,
                      float* __restrict__ o, size_t n) {
  GS_LOOP(i, n) o[i] = a[i] + b[i];
}

__global__ void k_negexp(const float* __restrict__ a, float* __restrict__ o, size_t n) {
  GS_LOOP(i, n) o[i] = -__expf(a[i]);
}

// one wave32 per row
__global__ void k_layernorm(const float* __restrict__ x, const float* __restrict__ w,
                            const float* __restrict__ b, float* __restrict__ y,
                            int M, int E) {
  int row = blockIdx.x * 8 + (threadIdx.x >> 5);
  int lane = threadIdx.x & 31;
  if (row >= M) return;
  const float* xr = x + (size_t)row * E;
  float s = 0.f, sq = 0.f;
  for (int i = lane; i < E; i += 32) { float v = xr[i]; s += v; sq += v * v; }
  for (int off = 16; off > 0; off >>= 1) {
    s  += __shfl_xor(s, off, 32);
    sq += __shfl_xor(sq, off, 32);
  }
  float mu = s / E;
  float var = sq / E - mu * mu;
  float rs = rsqrtf(var + 1e-5f);
  float* yr = y + (size_t)row * E;
  for (int i = lane; i < E; i += 32) yr[i] = (xr[i] - mu) * rs * w[i] + b[i];
}

// causal depthwise conv over L (D_CONV=4) + SiLU; xi lives interleaved in xz [L, 2*di]
__global__ void k_conv1d_silu(const float* __restrict__ xz, const float* __restrict__ cw,
                              const float* __restrict__ cb, float* __restrict__ xc,
                              int di, size_t total) {
  GS_LOOP(i, total) {
    int d = (int)(i % di);
    long l = (long)(i / di);
    float sacc = cb[d];
#pragma unroll
    for (int k = 0; k < 4; ++k) {
      long ls = l + k - 3;
      if (ls >= 0) sacc += xz[(size_t)ls * (size_t)(2 * di) + d] * cw[d * 4 + k];
    }
    xc[i] = d_silu(sacc);
  }
}

// selective scan: one lane per channel, 16-element state held in VGPRs
__global__ void k_scan(const float* __restrict__ u, const float* __restrict__ delta,
                       const float* __restrict__ Amat, const float* __restrict__ xdbl,
                       int xstride, int boff, int coff, const float* __restrict__ Dvec,
                       float* __restrict__ y, int L, int di) {
  int d = blockIdx.x * blockDim.x + threadIdx.x;
  if (d >= di) return;
  float h[16], Ad[16];
#pragma unroll
  for (int s = 0; s < 16; ++s) { h[s] = 0.f; Ad[s] = Amat[(size_t)d * 16 + s]; }
  float Dd = Dvec[d];
  for (int t = 0; t < L; ++t) {
    const float* Bt = xdbl + (size_t)t * xstride + boff;
    const float* Ct = xdbl + (size_t)t * xstride + coff;
    if (t + 1 < L) __builtin_prefetch(xdbl + (size_t)(t + 1) * xstride + boff, 0, 0);
    float dt = delta[(size_t)t * di + d];
    float ut = u[(size_t)t * di + d];
    float dtu = dt * ut;
    float yt = 0.f;
#pragma unroll
    for (int s = 0; s < 16; ++s) {
      h[s] = __expf(dt * Ad[s]) * h[s] + dtu * Bt[s];
      yt += h[s] * Ct[s];
    }
    y[(size_t)t * di + d] = yt + ut * Dd;
  }
}

__global__ void k_gate(float* __restrict__ y, const float* __restrict__ xz, int di,
                       size_t total) {
  GS_LOOP(i, total) {
    int d = (int)(i % di);
    size_t l = i / di;
    y[i] *= d_silu(xz[l * (size_t)(2 * di) + di + d]);
  }
}

// pixel shuffle: g[m][(o*N+i)*N+j] -> out[o][t][h*N+i][w*N+j]
__global__ void k_upnx_scatter(const float* __restrict__ g, float* __restrict__ out,
                               int T, int H, int W, int od, int N, size_t total) {
  int Cc = od * N * N;
  GS_LOOP(i, total) {
    size_t m = i / Cc;
    int c = (int)(i % Cc);
    int j = c % N, ii = (c / N) % N, o = c / (N * N);
    int w = (int)(m % W), h = (int)((m / W) % H), t = (int)(m / ((size_t)W * H));
    out[(((size_t)o * T + t) * ((size_t)H * N) + (size_t)h * N + ii) * ((size_t)W * N) +
        (size_t)w * N + j] = g[i];
  }
}

// act stored [H*W][C]; col [H*W][C*9] with k = (c*3+di)*3+dj
__global__ void k_im2col3x3(const float* __restrict__ act, float* __restrict__ col,
                            int C, int H, int W, size_t total) {
  int K = C * 9;
  GS_LOOP(i, total) {
    size_t hw = i / K;
    int k = (int)(i % K);
    int dj = k % 3, di2 = (k / 3) % 3, c = k / 9;
    int h = (int)(hw / W), w = (int)(hw % W);
    int hs = h + di2 - 1, ws = w + dj - 1;
    col[i] = (hs >= 0 && hs < H && ws >= 0 && ws < W)
                 ? act[((size_t)hs * W + ws) * C + c]
                 : 0.0f;
  }
}

// convex upsample; flow [HW][2], mask [HW][576]; writes up (2,8H,8W) and flow (2,H,W)
__global__ void k_upsample(const float* __restrict__ flow, const float* __restrict__ mask,
                           float* __restrict__ up, float* __restrict__ flo, int H, int W,
                           size_t total) {
  GS_LOOP(idx, total) {
    int ij = (int)(idx & 63);
    int j = ij & 7, i2 = ij >> 3;
    size_t hw = idx >> 6;
    int h = (int)(hw / W), w = (int)(hw % W);
    float mv[9];
    float mx = -3.0e38f;
#pragma unroll
    for (int k = 0; k < 9; ++k) {
      mv[k] = mask[hw * 576 + (size_t)(k * 8 + i2) * 8 + j];
      mx = fmaxf(mx, mv[k]);
    }
    float se = 0.f;
#pragma unroll
    for (int k = 0; k < 9; ++k) { mv[k] = __expf(mv[k] - mx); se += mv[k]; }
    float inv = 1.0f / se, a0 = 0.f, a1 = 0.f;
#pragma unroll
    for (int k = 0; k < 9; ++k) {
      int ki = k / 3, kj = k % 3;
      int hs = h + ki - 1, ws = w + kj - 1;
      float f0 = 0.f, f1 = 0.f;
      if (hs >= 0 && hs < H && ws >= 0 && ws < W) {
        f0 = 8.0f * flow[((size_t)hs * W + ws) * 2 + 0];
        f1 = 8.0f * flow[((size_t)hs * W + ws) * 2 + 1];
      }
      float wt = mv[k] * inv;
      a0 += wt * f0;
      a1 += wt * f1;
    }
    size_t HO = (size_t)H * 8, WO = (size_t)W * 8;
    up[((size_t)h * 8 + i2) * WO + (size_t)w * 8 + j] = a0;
    up[HO * WO + ((size_t)h * 8 + i2) * WO + (size_t)w * 8 + j] = a1;
    if (ij == 0) {
      flo[hw] = flow[hw * 2];
      flo[(size_t)H * W + hw] = flow[hw * 2 + 1];
    }
  }
}

// ---------------- host orchestration ----------------
struct Reader {
  void* const* p;
  int i;
  const float* next() { return (const float*)p[i++]; }
};
struct BlockP {
  const float *ln_w, *ln_b, *in_w, *conv_w, *conv_b, *xproj_w, *dt_w, *dt_b, *A_log, *D, *out_w;
};
struct StageP {
  const float *patch_w, *patch_b, *tpos;
  BlockP blk[2];
  const float *nf_w, *nf_b;
};
static BlockP read_block(Reader& r) {
  BlockP b;
  b.ln_w = r.next(); b.ln_b = r.next(); b.in_w = r.next(); b.conv_w = r.next();
  b.conv_b = r.next(); b.xproj_w = r.next(); b.dt_w = r.next(); b.dt_b = r.next();
  b.A_log = r.next(); b.D = r.next(); b.out_w = r.next();
  return b;
}
static StageP read_stage(Reader& r, bool tpos) {
  StageP s;
  s.patch_w = r.next(); s.patch_b = r.next();
  s.tpos = tpos ? r.next() : nullptr;
  s.blk[0] = read_block(r); s.blk[1] = read_block(r);
  s.nf_w = r.next(); s.nf_b = r.next();
  return s;
}

struct WS {
  char* base;
  size_t off, cap;
  float* alloc(size_t n) {
    size_t b = (n * 4 + 255) & ~(size_t)255;
    float* p = (float*)(base + off);
    off += b;
    return p;
  }
};

static inline dim3 ew_grid(size_t total) {
  size_t b = (total + 255) / 256;
  if (b > 32768) b = 32768;
  if (b == 0) b = 1;
  return dim3((unsigned)b);
}
static inline int cdiv(int a, int b) { return (a + b - 1) / b; }

static void gemm(hipStream_t s, const float* A, const float* B, const float* bias, float* C,
                 int M, int N, int K, int lda, int ldc, int act, float scale) {
  dim3 g(cdiv(N, BN), cdiv(M, BM));
  k_wmma_gemm<<<g, 256, 0, s>>>(A, B, bias, C, M, N, K, lda, ldc, act, scale);
}
static void ln(hipStream_t s, const float* x, const float* w, const float* b, float* y, int M,
               int E) {
  k_layernorm<<<cdiv(M, 8), 256, 0, s>>>(x, w, b, y, M, E);
}
static void addv(hipStream_t s, const float* a, const float* b, float* o, size_t n) {
  k_add<<<ew_grid(n), 256, 0, s>>>(a, b, o, n);
}

static void run_mixer(hipStream_t s, WS& ws, const BlockP& bp, const float* xln, float* out,
                      int L, int E) {
  int di = 2 * E, R = (E + 15) / 16, xs = R + 32;
  float* xz    = ws.alloc((size_t)L * 2 * di);
  float* xc    = ws.alloc((size_t)L * di);
  float* xdbl  = ws.alloc((size_t)L * xs);
  float* delta = ws.alloc((size_t)L * di);
  float* Amat  = ws.alloc((size_t)di * 16);
  float* y     = ws.alloc((size_t)L * di);
  gemm(s, xln, bp.in_w, nullptr, xz, L, 2 * di, E, E, 2 * di, 0, 1.f);
  k_conv1d_silu<<<ew_grid((size_t)L * di), 256, 0, s>>>(xz, bp.conv_w, bp.conv_b, xc, di,
                                                        (size_t)L * di);
  gemm(s, xc, bp.xproj_w, nullptr, xdbl, L, xs, di, di, xs, 0, 1.f);
  gemm(s, xdbl, bp.dt_w, bp.dt_b, delta, L, di, R, xs, di, 3 /*softplus*/, 1.f);
  k_negexp<<<ew_grid((size_t)di * 16), 256, 0, s>>>(bp.A_log, Amat, (size_t)di * 16);
  k_scan<<<cdiv(di, 256), 256, 0, s>>>(xc, delta, Amat, xdbl, xs, R, R + 16, bp.D, y, L, di);
  k_gate<<<ew_grid((size_t)L * di), 256, 0, s>>>(y, xz, di, (size_t)L * di);
  gemm(s, y, bp.out_w, nullptr, out, L, E, di, di, E, 0, 1.f);
}

static float* run_stage(hipStream_t s, WS& ws, const StageP& sp, const float* x, int C, int T,
                        int H, int W, int kt, int ps, int E) {
  int To = T / kt, Ho = H / ps, Wo = W / ps;
  int L = To * Ho * Wo;
  int K = C * kt * ps * ps;
  float* Ap  = ws.alloc((size_t)L * K);
  float* seq = ws.alloc((size_t)L * E);
  k_pack_patch<<<ew_grid((size_t)L * K), 256, 0, s>>>(x, Ap, C, T, H, W, kt, ps, To, Ho, Wo,
                                                      (size_t)L * K);
  gemm(s, Ap, sp.patch_w, sp.patch_b, seq, L, E, K, K, E, 0, 1.f);
  if (sp.tpos)
    k_add_tpos<<<ew_grid((size_t)L * E), 256, 0, s>>>(seq, sp.tpos, Ho * Wo, E, (size_t)L * E);
  float* xln   = ws.alloc((size_t)L * E);
  float* mixa  = ws.alloc((size_t)L * E);
  float* resid = ws.alloc((size_t)L * E);
  float* mixb  = ws.alloc((size_t)L * E);
  float* h     = ws.alloc((size_t)L * E);
  // block 0: residual = seq
  ln(s, seq, sp.blk[0].ln_w, sp.blk[0].ln_b, xln, L, E);
  run_mixer(s, ws, sp.blk[0], xln, mixa, L, E);
  // block 1: residual = mixa + seq
  addv(s, mixa, seq, resid, (size_t)L * E);
  ln(s, resid, sp.blk[1].ln_w, sp.blk[1].ln_b, xln, L, E);
  run_mixer(s, ws, sp.blk[1], xln, mixb, L, E);
  // final residual + norm_f
  addv(s, resid, mixb, resid, (size_t)L * E);
  ln(s, resid, sp.nf_w, sp.nf_b, h, L, E);
  return h;  // [L, E] == [(To,Ho,Wo), E]
}

extern "C" void kernel_launch(void* const* d_in, const int* in_sizes, int n_in, void* d_out,
                              int out_size, void* d_ws, size_t ws_size, hipStream_t stream) {
  (void)in_sizes; (void)n_in; (void)out_size;
  Reader r{d_in, 0};
  const float* x0 = r.next();  // (1,32,480,640) -> treated as (C=1,T=32,H=480,W=640)
  StageP mm1 = read_stage(r, true);
  const float* up1_w = r.next(); const float* up1_b = r.next();
  StageP mm2 = read_stage(r, true);
  const float* up2_w = r.next(); const float* up2_b = r.next();
  StageP mm3 = read_stage(r, true);
  const float* up3_w = r.next(); const float* up3_b = r.next();
  StageP mm4 = read_stage(r, false);
  const float* flow_w1 = r.next(); const float* flow_b1 = r.next();
  const float* flow_w2 = r.next(); const float* flow_b2 = r.next();
  const float* mask_w1 = r.next(); const float* mask_b1 = r.next();
  const float* mask_w2 = r.next(); const float* mask_b2 = r.next();

  WS ws{(char*)d_ws, 0, ws_size};

  // ---- stage 1 + up1 (N=16, out=32) -> x1 (32, 32, 240, 320)
  float* x1 = ws.alloc((size_t)32 * 32 * 240 * 320);
  {
    size_t mk = ws.off;
    float* h1 = run_stage(stream, ws, mm1, x0, 1, 32, 480, 640, 1, 32, 32);  // [9600,32]
    float* g = ws.alloc((size_t)9600 * 8192);
    gemm(stream, h1, up1_w, up1_b, g, 9600, 8192, 32, 32, 8192, 0, 1.f);
    k_upnx_scatter<<<ew_grid((size_t)9600 * 8192), 256, 0, stream>>>(
        g, x1, 32, 15, 20, 32, 16, (size_t)9600 * 8192);
    ws.off = mk;
  }
  // ---- stage 2 + up2 (N=4, out=64) -> x2 (64, 8, 120, 160)
  float* x2 = ws.alloc((size_t)64 * 8 * 120 * 160);
  {
    size_t mk = ws.off;
    float* h2 = run_stage(stream, ws, mm2, x1, 32, 32, 240, 320, 4, 8, 64);  // [9600,64]
    float* g = ws.alloc((size_t)9600 * 1024);
    gemm(stream, h2, up2_w, up2_b, g, 9600, 1024, 64, 64, 1024, 0, 1.f);
    k_upnx_scatter<<<ew_grid((size_t)9600 * 1024), 256, 0, stream>>>(
        g, x2, 8, 30, 40, 64, 4, (size_t)9600 * 1024);
    ws.off = mk;
  }
  // ---- stage 3 + up3 (N=2, out=128) -> x3 (128, 4, 60, 80)
  float* x3 = ws.alloc((size_t)128 * 4 * 60 * 80);
  {
    size_t mk = ws.off;
    float* h3 = run_stage(stream, ws, mm3, x2, 64, 8, 120, 160, 2, 4, 256);  // [4800,256]
    float* g = ws.alloc((size_t)4800 * 512);
    gemm(stream, h3, up3_w, up3_b, g, 4800, 512, 256, 256, 512, 0, 1.f);
    k_upnx_scatter<<<ew_grid((size_t)4800 * 512), 256, 0, stream>>>(
        g, x3, 4, 30, 40, 128, 2, (size_t)4800 * 512);
    ws.off = mk;
  }
  // ---- stage 4 + flow/mask head
  {
    float* h4 = run_stage(stream, ws, mm4, x3, 128, 4, 60, 80, 4, 1, 512);  // [4800,512]
    const int HW = 4800, H = 60, W = 80;
    float* col = ws.alloc((size_t)HW * 512 * 9);
    k_im2col3x3<<<ew_grid((size_t)HW * 512 * 9), 256, 0, stream>>>(h4, col, 512, H, W,
                                                                   (size_t)HW * 512 * 9);
    float* f = ws.alloc((size_t)HW * 256);
    gemm(stream, col, flow_w1, flow_b1, f, HW, 256, 4608, 4608, 256, 2 /*relu*/, 1.f);
    float* m = ws.alloc((size_t)HW * 256);
    gemm(stream, col, mask_w1, mask_b1, m, HW, 256, 4608, 4608, 256, 2 /*relu*/, 1.f);
    float* col2 = ws.alloc((size_t)HW * 256 * 9);
    k_im2col3x3<<<ew_grid((size_t)HW * 256 * 9), 256, 0, stream>>>(f, col2, 256, H, W,
                                                                   (size_t)HW * 256 * 9);
    float* flow = ws.alloc((size_t)HW * 2);
    gemm(stream, col2, flow_w2, flow_b2, flow, HW, 2, 2304, 2304, 2, 0, 1.f);
    float* upmask = ws.alloc((size_t)HW * 576);
    gemm(stream, m, mask_w2, mask_b2, upmask, HW, 576, 256, 256, 576, 0, 0.5f);
    float* out_up = (float*)d_out;                          // (2, 480, 640)
    float* out_fl = (float*)d_out + (size_t)2 * 480 * 640;  // (2, 60, 80)
    k_upsample<<<ew_grid((size_t)HW * 64), 256, 0, stream>>>(flow, upmask, out_up, out_fl, H,
                                                             W, (size_t)HW * 64);
  }
}
